// MultiHeadAttention_41850161332323
// MI455X (gfx1250) — compile-verified, matrix operand order checked by reading, not run on hardware
//
#include <hip/hip_runtime.h>
#include <stdint.h>

// ---------------- problem constants ----------------
#define Bb   2
#define SEQ  2048
#define DM   1024
#define NH   16
#define HD   64
#define MROWS (Bb*SEQ)   // 4096 rows for all [B*S, D] GEMMs

// ---------------- WMMA types ----------------
typedef __attribute__((ext_vector_type(16))) __bf16 v16bf;
typedef __attribute__((ext_vector_type(8)))  float  v8f;

union Frag16 { uint32_t u[8]; v16bf v; };
union U4u16  { uint4 q; unsigned short s[8]; uint32_t u[4]; };

__device__ __forceinline__ unsigned short f2bf(float f) {
    // round-to-nearest-even fp32 -> bf16
    uint32_t u = __float_as_uint(f);
    uint32_t r = u + 0x7FFFu + ((u >> 16) & 1u);
    return (unsigned short)(r >> 16);
}

// LDS byte offset (within workgroup allocation) of a generic pointer into shared mem
__device__ __forceinline__ uint32_t lds_off(const void* p) {
    return (uint32_t)(uintptr_t)(__attribute__((address_space(3))) const void*)p;
}

// CDNA5 async DMA: global -> LDS, 16B per lane, tracked by ASYNCcnt
__device__ __forceinline__ void async_load_b128(uint32_t ldsaddr, const void* gptr) {
    asm volatile("global_load_async_to_lds_b128 %0, %1, off"
                 :: "v"(ldsaddr), "v"(gptr) : "memory");
}
__device__ __forceinline__ void wait_async0() {
    asm volatile("s_wait_asynccnt 0x0" ::: "memory");
}
__device__ __forceinline__ void wait_ds0() {
    asm volatile("s_wait_dscnt 0x0" ::: "memory");
}

// ---------------- fp32 -> bf16 convert ----------------
__global__ __launch_bounds__(256)
void cvt_f32_bf16(const float* __restrict__ x, unsigned short* __restrict__ y, int n) {
    int i = (blockIdx.x * 256 + threadIdx.x) * 4;
    if (i < n) {
        float4 f = *(const float4*)(x + i);
        y[i + 0] = f2bf(f.x);
        y[i + 1] = f2bf(f.y);
        y[i + 2] = f2bf(f.z);
        y[i + 3] = f2bf(f.w);
    }
}

// ---------------- GEMM: C[M=4096,N=1024] = A(bf16) @ W(bf16) + bias ----------------
// BM=128 (8 waves x 16 rows), BN=64 (4 WMMA n-tiles), BK=32 (one WMMA k-step).
// A tile: async DMA to LDS, double-buffered.  W tile: register-prefetched, scatter-transposed.
// mode 0: write bf16 to per-head layout [B,H,S,HD]   (QKV projections)
// mode 1: write fp32 to flat [M,N]                   (final @ Wo -> d_out)
#define BM 128
#define BN 64
#define BK 32
#define PKA 40   // A row stride (ushorts): 80B = 16B-aligned for async B128, conflict-free
#define PKB 36   // Bst k-stride (ushorts): 72B, conflict-free for 16-lane gathers

__global__ __launch_bounds__(256)
void gemm_bf16(const unsigned short* __restrict__ A,
               const unsigned short* __restrict__ W,
               const float* __restrict__ bias,
               void* __restrict__ out, int mode)
{
    __shared__ unsigned short As[2][BM * PKA];   // [row][k]
    __shared__ unsigned short Bst[2][BN * PKB];  // [n][k] transposed

    const int tid  = threadIdx.x;
    const int wv   = tid >> 5;
    const int lane = tid & 31;
    const int lo   = lane & 15;
    const int hi   = lane >> 4;
    const int bm   = blockIdx.y * BM;
    const int bn   = blockIdx.x * BN;
    const int kb   = hi * 8;

    auto issueA = [&](int buf, int k0) {
#pragma unroll
        for (int c = tid; c < 512; c += 256) {           // 128 rows x 4 16B segs
            int row = c >> 2, seg = c & 3;
            async_load_b128(lds_off(&As[buf][row * PKA + seg * 8]),
                            A + (size_t)(bm + row) * DM + k0 + seg * 8);
        }
    };
    auto loadB = [&](int k0) {
        U4u16 d;
        int row = tid >> 3, seg = tid & 7;               // 32 rows x 8 16B segs
        d.q = *(const uint4*)(W + (size_t)(k0 + row) * DM + bn + seg * 8);
        return d;
    };
    auto storeB = [&](int buf, const U4u16& d) {
        int row = tid >> 3, seg = tid & 7;
#pragma unroll
        for (int j = 0; j < 8; ++j) Bst[buf][(seg * 8 + j) * PKB + row] = d.s[j];
    };

    v8f acc[4];
#pragma unroll
    for (int t = 0; t < 4; ++t)
#pragma unroll
        for (int e = 0; e < 8; ++e) acc[t][e] = 0.0f;

    // prologue: tile 0 in flight, W tile 1 in registers
    U4u16 breg = loadB(0);
    issueA(0, 0);
    storeB(0, breg);
    breg = loadB(BK);

    const int NIT = DM / BK;  // 32
    for (int kk = 0; kk < NIT; ++kk) {
        const int p = kk & 1;
        wait_async0();          // this wave's As[p] DMA complete
        __syncthreads();        // all waves' stores/DMAs visible; prev reads of buf 1-p done
        if (kk + 1 < NIT) {
            issueA(1 - p, (kk + 1) * BK);     // prefetch under this iteration's compute
            storeB(1 - p, breg);
            if (kk + 2 < NIT) breg = loadB((kk + 2) * BK);
        }

        // ---- WMMA fragments per ISA 16-bit layout ----
        Frag16 a;
        const int arow = wv * 16 + lo;
#pragma unroll
        for (int j = 0; j < 4; ++j) {
            a.u[j]     = *(const uint32_t*)&As[p][arow * PKA + kb + 2 * j];
            a.u[j + 4] = *(const uint32_t*)&As[p][arow * PKA + kb + 16 + 2 * j];
        }
#pragma unroll
        for (int t = 0; t < 4; ++t) {
            Frag16 b;
            const int ncol = t * 16 + lo;
#pragma unroll
            for (int j = 0; j < 4; ++j) {
                b.u[j]     = *(const uint32_t*)&Bst[p][ncol * PKB + kb + 2 * j];
                b.u[j + 4] = *(const uint32_t*)&Bst[p][ncol * PKB + kb + 16 + 2 * j];
            }
            acc[t] = __builtin_amdgcn_wmma_f32_16x16x32_bf16(
                false, a.v, false, b.v, (short)0, acc[t], false, false);
        }
    }

    // ---- epilogue: C/D layout M = e + 8*hi, N = t*16 + lo ----
#pragma unroll
    for (int t = 0; t < 4; ++t) {
        const int n = bn + t * 16 + lo;
        const float bs = bias[n];
#pragma unroll
        for (int e = 0; e < 8; ++e) {
            const int m = bm + wv * 16 + e + hi * 8;
            float val = acc[t][e] + bs;
            if (mode == 0) {
                int b = m >> 11, s = m & (SEQ - 1);
                int h = n >> 6,  d = n & (HD - 1);
                ((unsigned short*)out)[((size_t)(b * NH + h) * SEQ + s) * HD + d] = f2bf(val);
            } else {
                ((float*)out)[(size_t)m * DM + n] = val;
            }
        }
    }
}

// ---------------- Flash attention: per block = one (b,h) x 128 query rows ----------------
// 8 waves, each wave owns 16 query rows end-to-end (private online-softmax state).
// Q + K tiles: async DMA to LDS.  K/V double-buffered; V register-prefetched + transposed.
#define QP 72    // padded stride (ushorts): 144B = 16B-aligned for async B128, conflict-free

__global__ __launch_bounds__(256)
void attn_kernel(const unsigned short* __restrict__ Qp,
                 const unsigned short* __restrict__ Kp,
                 const unsigned short* __restrict__ Vp,
                 unsigned short* __restrict__ attnb)
{
    __shared__ unsigned short Qs[128 * QP];       // [qrow][d]
    __shared__ unsigned short Ks[2][64 * QP];     // [key][d]  (B-frag: col=key, contract=d)
    __shared__ unsigned short Vt[2][64 * QP];     // [d][key]  (B-frag: col=d, contract=key)
    __shared__ unsigned short Ps[8 * 16 * QP];    // per-wave P [16][key]

    const int tid  = threadIdx.x;
    const int wv   = tid >> 5;
    const int lane = tid & 31;
    const int lo   = lane & 15;
    const int hi   = lane >> 4;
    const int kb   = hi * 8;
    const int bh   = blockIdx.x;    // b*NH + h
    const int qblk = blockIdx.y;    // 0..15

    const size_t headBase = (size_t)bh * SEQ * HD;

    auto issueK = [&](int buf, int kt) {
#pragma unroll
        for (int c = tid; c < 512; c += 256) {           // 64 rows x 8 16B segs
            int row = c >> 3, seg = c & 7;
            async_load_b128(lds_off(&Ks[buf][row * QP + seg * 8]),
                            Kp + headBase + (size_t)(kt * 64 + row) * HD + seg * 8);
        }
    };
    auto loadV = [&](int kt, U4u16* vr) {
#pragma unroll
        for (int i = 0; i < 2; ++i) {
            int c = tid + i * 256;
            int row = c >> 3, seg = c & 7;
            vr[i].q = *(const uint4*)(Vp + headBase + (size_t)(kt * 64 + row) * HD + seg * 8);
        }
    };
    auto storeV = [&](int buf, const U4u16* vr) {
#pragma unroll
        for (int i = 0; i < 2; ++i) {
            int c = tid + i * 256;
            int row = c >> 3, seg = c & 7;
#pragma unroll
            for (int j = 0; j < 8; ++j) Vt[buf][(seg * 8 + j) * QP + row] = vr[i].s[j];
        }
    };

    // ---- prologue: Q tile + tile 0 in flight ----
#pragma unroll
    for (int c = tid; c < 1024; c += 256) {              // 128 rows x 8 16B segs
        int row = c >> 3, seg = c & 7;
        async_load_b128(lds_off(&Qs[row * QP + seg * 8]),
                        Qp + headBase + (size_t)(qblk * 128 + row) * HD + seg * 8);
    }
    U4u16 vr[2];
    loadV(0, vr);
    issueK(0, 0);
    storeV(0, vr);
    loadV(1, vr);

    float mrow[8], lrow[8];
#pragma unroll
    for (int e = 0; e < 8; ++e) { mrow[e] = -3.0e38f; lrow[e] = 0.0f; }
    v8f accO[4];
#pragma unroll
    for (int t = 0; t < 4; ++t)
#pragma unroll
        for (int e = 0; e < 8; ++e) accO[t][e] = 0.0f;

    const float scale = 0.125f; // 1/sqrt(64)
    const int NT = SEQ / 64;    // 32

    for (int kt = 0; kt < NT; ++kt) {
        const int p = kt & 1;
        wait_async0();          // Q (first iter) + Ks[p] DMA complete for this wave
        __syncthreads();        // all waves' tiles visible; prior reads of buf 1-p done
        if (kt + 1 < NT) {
            issueK(1 - p, kt + 1);
            storeV(1 - p, vr);
            if (kt + 2 < NT) loadV(kt + 2, vr);
        }

        // ---- S = Q @ K^T (16 rows x 64 keys per wave), contract over HD=64 ----
        v8f sc[4];
#pragma unroll
        for (int t = 0; t < 4; ++t)
#pragma unroll
            for (int e = 0; e < 8; ++e) sc[t][e] = 0.0f;

#pragma unroll
        for (int ks = 0; ks < 2; ++ks) {
            Frag16 a;
            const int arow = wv * 16 + lo;
#pragma unroll
            for (int j = 0; j < 4; ++j) {
                a.u[j]     = *(const uint32_t*)&Qs[arow * QP + ks * 32 + kb + 2 * j];
                a.u[j + 4] = *(const uint32_t*)&Qs[arow * QP + ks * 32 + kb + 16 + 2 * j];
            }
#pragma unroll
            for (int t = 0; t < 4; ++t) {
                Frag16 b;
                const int key = t * 16 + lo;
#pragma unroll
                for (int j = 0; j < 4; ++j) {
                    b.u[j]     = *(const uint32_t*)&Ks[p][key * QP + ks * 32 + kb + 2 * j];
                    b.u[j + 4] = *(const uint32_t*)&Ks[p][key * QP + ks * 32 + kb + 16 + 2 * j];
                }
                sc[t] = __builtin_amdgcn_wmma_f32_16x16x32_bf16(
                    false, a.v, false, b.v, (short)0, sc[t], false, false);
            }
        }

        // ---- online softmax: row r = e + 8*hi lives in 16 lanes of a half-wave ----
        float pm[8];
#pragma unroll
        for (int e = 0; e < 8; ++e) {
            float v0 = fmaxf(fmaxf(sc[0][e], sc[1][e]), fmaxf(sc[2][e], sc[3][e])) * scale;
#pragma unroll
            for (int off = 1; off < 16; off <<= 1)
                v0 = fmaxf(v0, __shfl_xor(v0, off, 32));
            pm[e] = v0;
        }
        float alpha[8], rs[8];
#pragma unroll
        for (int e = 0; e < 8; ++e) {
            float mn = fmaxf(mrow[e], pm[e]);
            alpha[e] = __expf(mrow[e] - mn);
            mrow[e]  = mn;
            rs[e]    = 0.0f;
        }
#pragma unroll
        for (int t = 0; t < 4; ++t) {
#pragma unroll
            for (int e = 0; e < 8; ++e) {
                float pvl = __expf(sc[t][e] * scale - mrow[e]);
                rs[e] += pvl;
                Ps[(wv * 16 + e + hi * 8) * QP + t * 16 + lo] = f2bf(pvl);
                accO[t][e] *= alpha[e];
            }
        }
#pragma unroll
        for (int e = 0; e < 8; ++e) {
            float r = rs[e];
#pragma unroll
            for (int off = 1; off < 16; off <<= 1)
                r += __shfl_xor(r, off, 32);
            lrow[e] = lrow[e] * alpha[e] + r;
        }

        // in-wave LDS RAW: P D-layout stores -> A-layout loads
        wait_ds0();

        // ---- O += P @ V, contract over 64 keys ----
#pragma unroll
        for (int ks = 0; ks < 2; ++ks) {
            Frag16 a;
#pragma unroll
            for (int j = 0; j < 4; ++j) {
                a.u[j]     = *(const uint32_t*)&Ps[(wv * 16 + lo) * QP + ks * 32 + kb + 2 * j];
                a.u[j + 4] = *(const uint32_t*)&Ps[(wv * 16 + lo) * QP + ks * 32 + kb + 16 + 2 * j];
            }
#pragma unroll
            for (int t = 0; t < 4; ++t) {
                Frag16 b;
                const int dcol = t * 16 + lo;
#pragma unroll
                for (int j = 0; j < 4; ++j) {
                    b.u[j]     = *(const uint32_t*)&Vt[p][dcol * QP + ks * 32 + kb + 2 * j];
                    b.u[j + 4] = *(const uint32_t*)&Vt[p][dcol * QP + ks * 32 + kb + 16 + 2 * j];
                }
                accO[t] = __builtin_amdgcn_wmma_f32_16x16x32_bf16(
                    false, a.v, false, b.v, (short)0, accO[t], false, false);
            }
        }
    }

    // ---- normalize + write attn output (bf16, [B,S,D] with heads concat) ----
    const int b = bh >> 4, h = bh & 15;
#pragma unroll
    for (int t = 0; t < 4; ++t) {
#pragma unroll
        for (int e = 0; e < 8; ++e) {
            const int q = qblk * 128 + wv * 16 + e + hi * 8;
            const int d = t * 16 + lo;
            float o = accO[t][e] / lrow[e];
            attnb[(size_t)(b * SEQ + q) * DM + h * HD + d] = f2bf(o);
        }
    }
}

// ---------------- host side ----------------
extern "C" void kernel_launch(void* const* d_in, const int* in_sizes, int n_in,
                              void* d_out, int out_size, void* d_ws, size_t ws_size,
                              hipStream_t stream)
{
    (void)in_sizes; (void)n_in; (void)out_size; (void)ws_size;

    const float* q  = (const float*)d_in[0];
    const float* k  = (const float*)d_in[1];
    const float* v  = (const float*)d_in[2];
    const float* Wq = (const float*)d_in[3];
    const float* Wk = (const float*)d_in[4];
    const float* Wv = (const float*)d_in[5];
    const float* Wo = (const float*)d_in[6];
    const float* bq = (const float*)d_in[7];
    const float* bk = (const float*)d_in[8];
    const float* bv = (const float*)d_in[9];
    const float* bo = (const float*)d_in[10];

    const size_t ACT = (size_t)MROWS * DM;  // 4 Mi elements
    const size_t WEL = (size_t)DM * DM;     // 1 Mi elements

    unsigned short* qb    = (unsigned short*)d_ws;
    unsigned short* kbuf  = qb   + ACT;
    unsigned short* vbuf  = kbuf + ACT;
    unsigned short* wqb   = vbuf + ACT;
    unsigned short* wkb   = wqb  + WEL;
    unsigned short* wvb   = wkb  + WEL;
    unsigned short* wob   = wvb  + WEL;
    unsigned short* Qp    = wob  + WEL;   // [B,H,S,HD] bf16
    unsigned short* Kp    = Qp   + ACT;
    unsigned short* Vp    = Kp   + ACT;
    unsigned short* attnb = Vp   + ACT;   // [B,S,D] bf16
    // total: 7*ACT + 4*WEL ushorts = 64 MiB of workspace

    // 1) convert to bf16
    cvt_f32_bf16<<<(int)(ACT / 1024), 256, 0, stream>>>(q,  qb,   (int)ACT);
    cvt_f32_bf16<<<(int)(ACT / 1024), 256, 0, stream>>>(k,  kbuf, (int)ACT);
    cvt_f32_bf16<<<(int)(ACT / 1024), 256, 0, stream>>>(v,  vbuf, (int)ACT);
    cvt_f32_bf16<<<(int)(WEL / 1024), 256, 0, stream>>>(Wq, wqb,  (int)WEL);
    cvt_f32_bf16<<<(int)(WEL / 1024), 256, 0, stream>>>(Wk, wkb,  (int)WEL);
    cvt_f32_bf16<<<(int)(WEL / 1024), 256, 0, stream>>>(Wv, wvb,  (int)WEL);
    cvt_f32_bf16<<<(int)(WEL / 1024), 256, 0, stream>>>(Wo, wob,  (int)WEL);

    // 2) QKV projections -> per-head bf16
    dim3 gg(DM / BN, MROWS / BM);  // (16, 32)
    gemm_bf16<<<gg, 256, 0, stream>>>(qb,   wqb, bq, Qp, 0);
    gemm_bf16<<<gg, 256, 0, stream>>>(kbuf, wkb, bk, Kp, 0);
    gemm_bf16<<<gg, 256, 0, stream>>>(vbuf, wvb, bv, Vp, 0);

    // 3) flash attention
    attn_kernel<<<dim3(Bb * NH, SEQ / 128), 256, 0, stream>>>(Qp, Kp, Vp, attnb);

    // 4) output projection -> fp32 d_out
    gemm_bf16<<<gg, 256, 0, stream>>>(attnb, wob, bo, d_out, 1);
}
